// MPNN_55405078119119
// MI455X (gfx1250) — matrix operand order repaired and checked by reading, not compiled, not run on hardware
//
#include <hip/hip_runtime.h>

#define DEVINL __device__ __forceinline__

typedef __attribute__((ext_vector_type(16))) __bf16 v16bf;
typedef __attribute__((ext_vector_type(8)))  float  v8f;

constexpr int kNodes  = 10000;
constexpr int kEdges  = 64000;
constexpr int kGraphs = 64;
constexpr int KW2     = 64 * 64 + 64;   // 4160 contraction rows (w2 + bias rows)

// --- scalar f32 -> bf16 (RNE, cold paths only)
DEVINL __bf16 f2bf(float f) {
  unsigned u = __builtin_bit_cast(unsigned, f);
  u += 0x7fffu + ((u >> 16) & 1u);
  unsigned short h = (unsigned short)(u >> 16);
  return __builtin_bit_cast(__bf16, h);
}

// --- pack two f32 into a bf16x2 dword by truncation: one v_perm_b32
DEVINL unsigned pack2bf(float f0, float f1) {
  return __builtin_amdgcn_perm(__builtin_bit_cast(unsigned, f1),
                               __builtin_bit_cast(unsigned, f0), 0x07060302u);
}

// --- packed bf16 multiply (V_PK_MUL_BF16, VOP3P op 42)
DEVINL unsigned pkmul_bf16(unsigned a, unsigned b) {
  unsigned d;
  asm("v_pk_mul_bf16 %0, %1, %2" : "=v"(d) : "v"(a), "v"(b));
  return d;
}

DEVINL float lrelu(float v) { return v > 0.0f ? v : 0.01f * v; }

DEVINL v8f wmma_bf16(v16bf a, v16bf b, v8f c) {
  return __builtin_amdgcn_wmma_f32_16x16x32_bf16(false, a, false, b, (short)0, c,
                                                 false, false);
}

// 16 consecutive bf16 (32B, 16B-aligned) -> v16bf
DEVINL v16bf load_bfrag16(const __bf16* p) {
  union { v16bf v; uint4 q[2]; } u;
  u.q[0] = *(const uint4*)(p);
  u.q[1] = *(const uint4*)(p + 8);
  return u.v;
}

// B stored transposed: Bt[n * ldK + c]. B layout: lane = column, koff = 0 (lanes<16) / 8 (lanes>=16);
// elements [0..7] = K = c0+koff+j, [8..15] = K = c0+16+koff+j (matches A).
DEVINL v16bf bfragT(const __bf16* Bt, int ldK, int n_abs, int c0, int lane) {
  int koff = (lane & 16) ? 8 : 0;
  const __bf16* row = Bt + (size_t)n_abs * ldK;
  union { v16bf v; uint4 q[2]; } u;
  u.q[0] = *(const uint4*)(row + c0 + koff);
  u.q[1] = *(const uint4*)(row + c0 + 16 + koff);
  return u.v;
}

// ISA A layout (16-bit 16x32): lane<16 -> koff 0, lane>=16 -> koff 8;
// elements [0..7] = K = k0+koff+j, [8..15] = K = k0+16+koff+j.
DEVINL v16bf afrag_bf16row(const __bf16* row, int k0, int koff) {
  union { v16bf v; uint4 q[2]; } u;
  u.q[0] = *(const uint4*)(row + k0 + koff);
  u.q[1] = *(const uint4*)(row + k0 + 16 + koff);
  return u.v;
}

// A-frag from an f32 row, truncation-packed with v_perm_b32.
DEVINL v16bf afrag_f32row(const float* rowA, int k0, int lane) {
  int koff = (lane & 16) ? 8 : 0;
  const float4* p0 = (const float4*)(rowA + k0 + koff);
  const float4* p1 = (const float4*)(rowA + k0 + 16 + koff);
  float4 a0 = p0[0], a1 = p0[1], b0 = p1[0], b1 = p1[1];
  union { v16bf v; unsigned q[8]; } u;
  u.q[0] = pack2bf(a0.x, a0.y); u.q[1] = pack2bf(a0.z, a0.w);
  u.q[2] = pack2bf(a1.x, a1.y); u.q[3] = pack2bf(a1.z, a1.w);
  u.q[4] = pack2bf(b0.x, b0.y); u.q[5] = pack2bf(b0.z, b0.w);
  u.q[6] = pack2bf(b1.x, b1.y); u.q[7] = pack2bf(b1.z, b1.w);
  return u.v;
}

// scale a bf16 fragment by a broadcast bf16 pair: 8 x v_pk_mul_bf16
DEVINL v16bf scale_frag(v16bf h, unsigned tpair) {
  union { v16bf v; unsigned q[8]; } u, r;
  u.v = h;
#pragma unroll
  for (int i = 0; i < 8; ++i) r.q[i] = pkmul_bf16(u.q[i], tpair);
  return r.v;
}

// ---------------- weight transpose/convert: out[n*ldOut + colOff + k] = bf16(in[k*N+n])
__global__ void transpose_bf_kernel(const float* __restrict__ in, __bf16* __restrict__ out,
                                    int K, int N, int ldOut, int colOff) {
  int idx = blockIdx.x * blockDim.x + threadIdx.x;
  if (idx >= K * N) return;
  int k = idx / N, n = idx % N;
  out[(size_t)n * ldOut + colOff + k] = f2bf(in[idx]);
}

// ---------------- generic row-tile WMMA GEMM: y[m,0:64] = act(in[m,:K] @ W + bias (+ add))
// Wt: [64][K] bf16 transposed. ACT: 0 = relu, 1 = leaky relu. Optional f32 and bf16 outputs.
template <int K, int ACT, bool HASADD, bool WF32, bool WBF>
__global__ void gemm_rowtile_kernel(const float* __restrict__ in,
                                    const __bf16* __restrict__ Wt,
                                    const float* __restrict__ bias,
                                    const float* __restrict__ add,
                                    float* __restrict__ out,
                                    __bf16* __restrict__ outbf, int M) {
  int lane = threadIdx.x & 31;
  int wave = threadIdx.x >> 5;
  int tile = blockIdx.x * 4 + wave;
  int row0 = tile * 16;
  if (row0 >= M) return;
  int mrow = lane & 15;
  int arow = row0 + mrow;
  if (arow > M - 1) arow = M - 1;
  const float* rowA = in + (size_t)arow * K;

  const v8f vzero = {0, 0, 0, 0, 0, 0, 0, 0};
  v8f acc[4] = {vzero, vzero, vzero, vzero};

#pragma unroll
  for (int k0 = 0; k0 < K; k0 += 32) {
    v16bf b0 = bfragT(Wt, K, 0 * 16 + mrow, k0, lane);
    v16bf b1 = bfragT(Wt, K, 1 * 16 + mrow, k0, lane);
    v16bf b2 = bfragT(Wt, K, 2 * 16 + mrow, k0, lane);
    v16bf b3 = bfragT(Wt, K, 3 * 16 + mrow, k0, lane);
    v16bf a = afrag_f32row(rowA, k0, lane);
    acc[0] = wmma_bf16(a, b0, acc[0]);
    acc[1] = wmma_bf16(a, b1, acc[1]);
    acc[2] = wmma_bf16(a, b2, acc[2]);
    acc[3] = wmma_bf16(a, b3, acc[3]);
  }

  int rbase = (lane & 16) ? 8 : 0;   // f32 C layout: VGPR r -> row r (lo lanes) / r+8 (hi)
#pragma unroll
  for (int nt = 0; nt < 4; ++nt) {
    int col = nt * 16 + mrow;
    float bv = bias[col];
#pragma unroll
    for (int r = 0; r < 8; ++r) {
      int row = row0 + r + rbase;
      if (row < M) {
        float v = acc[nt][r] + bv;
        if (HASADD) v += add[(size_t)row * 64 + col];
        v = (ACT == 0) ? (v > 0.0f ? v : 0.0f) : lrelu(v);
        if (WF32) out[(size_t)row * 64 + col] = v;
        if (WBF) outbf[(size_t)row * 64 + col] = f2bf(v);
      }
    }
  }
}

// ---------------- fused NNConv message GEMM + scatter
// msg = (t ⊗ h_src) @ B with B = [w2 as 4096x64 ; b2 as 64x64], K = 4160.
// A generated in-register: A[e, k*64+i] = t[e,k] * h[src[e], i]; k=64 -> scale 1 (bias rows).
// Wave: 32 edges (2 M-tiles sharing B fragments) x 64 outputs; h fragments hoisted in bf16.
// __launch_bounds__(128,1): lift the 128-VGPR allocator cap so the ~150 live regs fit
// without the scratch spill/reload of the hoisted h fragments.
__global__ void __launch_bounds__(128, 1)
msg_kernel(const __bf16* __restrict__ hbf,   // [kNodes,64] bf16
           const __bf16* __restrict__ tbf,   // [kEdges,64] bf16
           const __bf16* __restrict__ Bt,    // [64][KW2] bf16 transposed
           const int* __restrict__ src,
           const int* __restrict__ dst,
           float* __restrict__ agg) {        // [kNodes,64] pre-zeroed
  int lane = threadIdx.x & 31;
  int wave = threadIdx.x >> 5;
  int tile = blockIdx.x * 4 + wave;
  int e0   = tile * 32;
  int mrow = lane & 15;
  int koff = (lane & 16) ? 8 : 0;

  int eA = e0 + mrow;
  int eB = e0 + 16 + mrow;
  const __bf16* hAp = hbf + (size_t)src[eA] * 64;
  const __bf16* hBp = hbf + (size_t)src[eB] * 64;
  const unsigned short* tA = (const unsigned short*)(tbf + (size_t)eA * 64);
  const unsigned short* tB = (const unsigned short*)(tbf + (size_t)eB * 64);

  // hoisted, unscaled h fragments: 4 x v16bf = 32 VGPRs
  v16bf hfA[2], hfB[2];
#pragma unroll
  for (int s = 0; s < 2; ++s) {
    hfA[s] = afrag_bf16row(hAp, s * 32, koff);
    hfB[s] = afrag_bf16row(hBp, s * 32, koff);
  }

  const v8f vzero = {0, 0, 0, 0, 0, 0, 0, 0};
  v8f acc0[4] = {vzero, vzero, vzero, vzero};
  v8f acc1[4] = {vzero, vzero, vzero, vzero};

  for (int k = 0; k < 64; ++k) {
    unsigned sa = tA[k], sb = tB[k];
    unsigned pa = sa | (sa << 16);
    unsigned pb = sb | (sb << 16);
    int c_base = k * 64;
#pragma unroll
    for (int s = 0; s < 2; ++s) {
      int c0 = c_base + s * 32;
      // all 4 n-tile B loads issued first (one clause); A scaling overlaps latency
      v16bf b0 = bfragT(Bt, KW2, 0 * 16 + mrow, c0, lane);
      v16bf b1 = bfragT(Bt, KW2, 1 * 16 + mrow, c0, lane);
      v16bf b2 = bfragT(Bt, KW2, 2 * 16 + mrow, c0, lane);
      v16bf b3 = bfragT(Bt, KW2, 3 * 16 + mrow, c0, lane);
      {
        v16bf a = scale_frag(hfA[s], pa);     // aA live only for this block
        acc0[0] = wmma_bf16(a, b0, acc0[0]);
        acc0[1] = wmma_bf16(a, b1, acc0[1]);
        acc0[2] = wmma_bf16(a, b2, acc0[2]);
        acc0[3] = wmma_bf16(a, b3, acc0[3]);
      }
      {
        v16bf a = scale_frag(hfB[s], pb);     // aB reuses the same registers
        acc1[0] = wmma_bf16(a, b0, acc1[0]);
        acc1[1] = wmma_bf16(a, b1, acc1[1]);
        acc1[2] = wmma_bf16(a, b2, acc1[2]);
        acc1[3] = wmma_bf16(a, b3, acc1[3]);
      }
    }
  }
  // bias rows (scale 1 -> unscaled fragments, zero VALU)
#pragma unroll
  for (int s = 0; s < 2; ++s) {
    int c0 = 4096 + s * 32;
    v16bf b0 = bfragT(Bt, KW2, 0 * 16 + mrow, c0, lane);
    v16bf b1 = bfragT(Bt, KW2, 1 * 16 + mrow, c0, lane);
    v16bf b2 = bfragT(Bt, KW2, 2 * 16 + mrow, c0, lane);
    v16bf b3 = bfragT(Bt, KW2, 3 * 16 + mrow, c0, lane);
    acc0[0] = wmma_bf16(hfA[s], b0, acc0[0]);
    acc0[1] = wmma_bf16(hfA[s], b1, acc0[1]);
    acc0[2] = wmma_bf16(hfA[s], b2, acc0[2]);
    acc0[3] = wmma_bf16(hfA[s], b3, acc0[3]);
    acc1[0] = wmma_bf16(hfB[s], b0, acc1[0]);
    acc1[1] = wmma_bf16(hfB[s], b1, acc1[1]);
    acc1[2] = wmma_bf16(hfB[s], b2, acc1[2]);
    acc1[3] = wmma_bf16(hfB[s], b3, acc1[3]);
  }

  int rbase = (lane & 16) ? 8 : 0;
#pragma unroll
  for (int mt = 0; mt < 2; ++mt) {
#pragma unroll
    for (int r = 0; r < 8; ++r) {
      int e = e0 + mt * 16 + r + rbase;
      int d = dst[e];
      float* dstrow = agg + (size_t)d * 64;
#pragma unroll
      for (int nt = 0; nt < 4; ++nt) {
        float v = (mt == 0) ? acc0[nt][r] : acc1[nt][r];
        unsafeAtomicAdd(dstrow + nt * 16 + mrow, v);
      }
    }
  }
}

// ---------------- global_add_pool
__global__ void pool_kernel(const float* __restrict__ h, const int* __restrict__ batch,
                            float* __restrict__ hg, int M) {
  int idx = blockIdx.x * blockDim.x + threadIdx.x;
  if (idx >= M * 64) return;
  int v = idx >> 6, c = idx & 63;
  unsafeAtomicAdd(&hg[(size_t)batch[v] * 64 + c], h[idx]);
}

// ---------------- graph head: lrelu(hg @ fc1 + b1) @ fc2 + b2
__global__ void head_kernel(const float* __restrict__ hg, const float* __restrict__ fc1w,
                            const float* __restrict__ fc1b, const float* __restrict__ fc2w,
                            const float* __restrict__ fc2b, float* __restrict__ out) {
  int g = threadIdx.x;
  if (g >= kGraphs) return;
  float o = fc2b[0];
  for (int j = 0; j < 32; ++j) {
    float z = fc1b[j];
    for (int i = 0; i < 64; ++i) z += hg[g * 64 + i] * fc1w[i * 32 + j];
    z = lrelu(z);
    o += z * fc2w[j];
  }
  out[g] = o;
}

extern "C" void kernel_launch(void* const* d_in, const int* in_sizes, int n_in,
                              void* d_out, int out_size, void* d_ws, size_t ws_size,
                              hipStream_t stream) {
  (void)in_sizes; (void)n_in; (void)out_size; (void)ws_size;
  const float* x      = (const float*)d_in[0];
  const int*   eidx   = (const int*)d_in[1];
  const float* eattr  = (const float*)d_in[2];
  const int*   batch  = (const int*)d_in[3];
  const float* nfc_w  = (const float*)d_in[4];
  const float* nfc_b  = (const float*)d_in[5];
  const float* e1w1   = (const float*)d_in[6];
  const float* e1b1   = (const float*)d_in[7];
  const float* e1w2   = (const float*)d_in[8];
  const float* e1b2   = (const float*)d_in[9];
  const float* g1root = (const float*)d_in[10];
  const float* g1bias = (const float*)d_in[11];
  const float* e2w1   = (const float*)d_in[12];
  const float* e2b1   = (const float*)d_in[13];
  const float* e2w2   = (const float*)d_in[14];
  const float* e2b2   = (const float*)d_in[15];
  const float* g2root = (const float*)d_in[16];
  const float* g2bias = (const float*)d_in[17];
  const float* fc1w   = (const float*)d_in[18];
  const float* fc1b   = (const float*)d_in[19];
  const float* fc2w   = (const float*)d_in[20];
  const float* fc2b   = (const float*)d_in[21];
  const int* src = eidx;
  const int* dst = eidx + kEdges;

  char* p = (char*)d_ws;
  auto carve = [&](size_t bytes) {
    char* r = p;
    p += (bytes + 255) & ~(size_t)255;
    return r;
  };
  float*  hA    = (float*)carve((size_t)kNodes * 64 * 4);
  float*  hB    = (float*)carve((size_t)kNodes * 64 * 4);
  float*  agg   = (float*)carve((size_t)kNodes * 64 * 4);
  float*  hg    = (float*)carve((size_t)kGraphs * 64 * 4);
  __bf16* hAbf  = (__bf16*)carve((size_t)kNodes * 64 * 2);
  __bf16* hBbf  = (__bf16*)carve((size_t)kNodes * 64 * 2);
  __bf16* tbf   = (__bf16*)carve(((size_t)kEdges * 64 + 64) * 2);
  __bf16* nfcwT = (__bf16*)carve((size_t)64 * 128 * 2);
  __bf16* w1T1  = (__bf16*)carve((size_t)64 * 32 * 2);
  __bf16* w1T2  = (__bf16*)carve((size_t)64 * 32 * 2);
  __bf16* rT1   = (__bf16*)carve((size_t)64 * 64 * 2);
  __bf16* rT2   = (__bf16*)carve((size_t)64 * 64 * 2);
  __bf16* w2T1  = (__bf16*)carve((size_t)64 * KW2 * 2);
  __bf16* w2T2  = (__bf16*)carve((size_t)64 * KW2 * 2);

  auto tlaunch = [&](const float* in, __bf16* out, int K, int N, int ld, int off) {
    int total = K * N;
    transpose_bf_kernel<<<(total + 255) / 256, 256, 0, stream>>>(in, out, K, N, ld, off);
  };
  tlaunch(nfc_w,  nfcwT, 128, 64, 128, 0);
  tlaunch(e1w1,   w1T1,   32, 64,  32, 0);
  tlaunch(e2w1,   w1T2,   32, 64,  32, 0);
  tlaunch(g1root, rT1,    64, 64,  64, 0);
  tlaunch(g2root, rT2,    64, 64,  64, 0);
  tlaunch(e1w2,   w2T1, 4096, 64, KW2, 0);
  tlaunch(e1b2,   w2T1,   64, 64, KW2, 4096);
  tlaunch(e2w2,   w2T2, 4096, 64, KW2, 0);
  tlaunch(e2b2,   w2T2,   64, 64, KW2, 4096);

  int nodeBlocks = (kNodes + 63) / 64;   // 4 waves/block, 16 rows/wave
  int edgeBlocks = kEdges / 64;

  // node encoder: hA (+bf16 copy) = lrelu(x @ nfc_w + b)
  gemm_rowtile_kernel<128, 1, false, true, true>
      <<<nodeBlocks, 128, 0, stream>>>(x, nfcwT, nfc_b, nullptr, hA, hAbf, kNodes);

  // ---- layer 1
  gemm_rowtile_kernel<32, 0, false, false, true>
      <<<edgeBlocks, 128, 0, stream>>>(eattr, w1T1, e1b1, nullptr, nullptr, tbf, kEdges);
  hipMemsetAsync(agg, 0, (size_t)kNodes * 64 * 4, stream);
  msg_kernel<<<kEdges / 128, 128, 0, stream>>>(hAbf, tbf, w2T1, src, dst, agg);
  gemm_rowtile_kernel<64, 1, true, true, true>
      <<<nodeBlocks, 128, 0, stream>>>(hA, rT1, g1bias, agg, hB, hBbf, kNodes);

  // ---- layer 2
  gemm_rowtile_kernel<32, 0, false, false, true>
      <<<edgeBlocks, 128, 0, stream>>>(eattr, w1T2, e2b1, nullptr, nullptr, tbf, kEdges);
  hipMemsetAsync(agg, 0, (size_t)kNodes * 64 * 4, stream);
  msg_kernel<<<kEdges / 128, 128, 0, stream>>>(hBbf, tbf, w2T2, src, dst, agg);
  gemm_rowtile_kernel<64, 1, true, true, false>
      <<<nodeBlocks, 128, 0, stream>>>(hB, rT2, g2bias, agg, hA, nullptr, kNodes);

  // ---- pool + head
  hipMemsetAsync(hg, 0, (size_t)kGraphs * 64 * 4, stream);
  pool_kernel<<<(kNodes * 64 + 255) / 256, 256, 0, stream>>>(hA, batch, hg, kNodes);
  head_kernel<<<1, 64, 0, stream>>>(hg, fc1w, fc1b, fc2w, fc2b, (float*)d_out);
}
